// RevenueGCN_40690520163147
// MI455X (gfx1250) — matrix operand order repaired and checked by reading, not compiled, not run on hardware
//
#include <hip/hip_runtime.h>
#include <hip/hip_bf16.h>
#include <stddef.h>

// ---------------------------------------------------------------------------
// Problem constants (match reference)
// ---------------------------------------------------------------------------
#define N_USERS  100000
#define N_ITEMS  100000
#define N_NODES  (N_USERS + N_ITEMS)
#define N_EDGES  4000000
#define USER_DIM 256
#define ITEM_DIM 128
#define COMMON   64
#define HIDDEN   64
#define OUT_DIM  32

typedef __attribute__((ext_vector_type(16))) _Float16 v16h;
typedef __attribute__((ext_vector_type(8)))  float    v8f;

// ---------------------------------------------------------------------------
// Pack fp32 weight W[K,N] (row-major) into f16 WMMA B-fragments.
// Tile index = ntile*KT + kt. Per tile: 32 lanes x 16 halves contiguous.
// Lane l: group g = l>>4 (K-half), column n = (l&15)+16*ntile,
// element j -> K = kt*32 + g*16 + j   (contiguous k order, matched by A)
// ---------------------------------------------------------------------------
__global__ void pack_wmma_b(const float* __restrict__ W, _Float16* __restrict__ Bp,
                            int N, int KT) {
    int tile  = blockIdx.x;          // ntile*KT + kt
    int kt    = tile % KT;
    int ntile = tile / KT;
    int lane  = threadIdx.x;         // 0..31
    int g     = lane >> 4;
    int n     = (lane & 15) + ntile * 16;
    int kbase = kt * 32 + g * 16;
    _Float16* dst = Bp + ((size_t)tile * 32 + lane) * 16;
#pragma unroll
    for (int j = 0; j < 16; ++j)
        dst[j] = (_Float16)W[(size_t)(kbase + j) * N + n];
}

// ---------------------------------------------------------------------------
// WMMA GEMM: C[M, NT*16] = A[M,K] (f32, cast to f16) @ Bpacked (+ bias)
// One wave per 16-row band: A fragment loaded & converted ONCE per K-tile,
// reused across all NT column tiles (NT back-to-back WMMAs).
// A fragment (contiguous k order, same as packed B): lane l (m = l&15, g = l>>4):
//   elems 0..15 -> K = kt*32 + 16*g + j
// -> one contiguous 16-float run = 4 x float4 loads.
// ---------------------------------------------------------------------------
template <int NT>
__global__ void __launch_bounds__(256)
wmma_gemm_f16(const float* __restrict__ A, const _Float16* __restrict__ Bp,
              const float* __restrict__ bias, float* __restrict__ C,
              int M, int K) {
    const int N     = NT * 16;
    const int KT    = K >> 5;
    const int bands = M >> 4;
    int wave = blockIdx.x * 8 + (threadIdx.x >> 5);
    if (wave >= bands) return;                 // wave-uniform: EXEC stays all-1s
    const int lane = threadIdx.x & 31;
    const int g    = lane >> 4;
    const int ml   = lane & 15;

    const float4* rowA =
        reinterpret_cast<const float4*>(A + (size_t)(wave * 16 + ml) * K);

    v8f c[NT];
#pragma unroll
    for (int nt = 0; nt < NT; ++nt) c[nt] = {};

    for (int kt = 0; kt < KT; ++kt) {
        float4 a0 = rowA[kt * 8 + 4 * g];
        float4 a1 = rowA[kt * 8 + 4 * g + 1];
        float4 a2 = rowA[kt * 8 + 4 * g + 2];
        float4 a3 = rowA[kt * 8 + 4 * g + 3];
        v16h a;
        a[0]  = (_Float16)a0.x; a[1]  = (_Float16)a0.y;
        a[2]  = (_Float16)a0.z; a[3]  = (_Float16)a0.w;
        a[4]  = (_Float16)a1.x; a[5]  = (_Float16)a1.y;
        a[6]  = (_Float16)a1.z; a[7]  = (_Float16)a1.w;
        a[8]  = (_Float16)a2.x; a[9]  = (_Float16)a2.y;
        a[10] = (_Float16)a2.z; a[11] = (_Float16)a2.w;
        a[12] = (_Float16)a3.x; a[13] = (_Float16)a3.y;
        a[14] = (_Float16)a3.z; a[15] = (_Float16)a3.w;

#pragma unroll
        for (int nt = 0; nt < NT; ++nt) {
            const v16h b = *reinterpret_cast<const v16h*>(
                Bp + (((size_t)nt * KT + kt) * 32 + lane) * 16);
            c[nt] = __builtin_amdgcn_wmma_f32_16x16x32_f16(
                        /*neg_a=*/false, a, /*neg_b=*/false, b,
                        /*c_mod=*/(short)0, c[nt],
                        /*reuse_a=*/false, /*reuse_b=*/false);
        }
    }

    // C/D layout: VGPR r -> row r (lanes 0-15) / row 8+r (lanes 16-31), col = ml
#pragma unroll
    for (int nt = 0; nt < NT; ++nt) {
        const int n  = nt * 16 + ml;
        const float bv = bias ? bias[n] : 0.0f;
#pragma unroll
        for (int r = 0; r < 8; ++r) {
            int row = wave * 16 + r + 8 * g;
            C[(size_t)row * N + n] = c[nt][r] + bv;
        }
    }
}

// ---------------------------------------------------------------------------
// Degree / normalization helpers
// ---------------------------------------------------------------------------
__global__ void fill_f32(float* __restrict__ p, float v, int n) {
    int i = blockIdx.x * blockDim.x + threadIdx.x;
    if (i < n) p[i] = v;
}

__global__ void deg_scatter(const int* __restrict__ dst, const float* __restrict__ ew,
                            float* __restrict__ deg, int E) {
    int i = blockIdx.x * blockDim.x + threadIdx.x;
    if (i < E) atomicAdd(&deg[dst[i]], ew[i]);
}

__global__ void rsqrt_inplace(float* __restrict__ p, int n) {
    int i = blockIdx.x * blockDim.x + threadIdx.x;
    if (i < n) {
        float d = p[i];
        p[i] = (d > 0.0f) ? rsqrtf(d) : 0.0f;
    }
}

// acc[node][c] = h[node][c] * dinv[node]^2   (self-loop message, norm = dinv*1*dinv)
__global__ void self_msg(const float* __restrict__ h, const float* __restrict__ dinv,
                         float* __restrict__ acc, int total, int lgC) {
    int i = blockIdx.x * blockDim.x + threadIdx.x;
    if (i < total) {
        float d = dinv[i >> lgC];
        acc[i] = h[i] * d * d;
    }
}

// Edge scatter: 2^(lgChunks) consecutive threads handle one edge, 4 cols each.
__global__ void edge_scatter(const float* __restrict__ h,
                             const int* __restrict__ src, const int* __restrict__ dst,
                             const float* __restrict__ ew, const float* __restrict__ dinv,
                             float* __restrict__ acc, int E, int C, int lgChunks) {
    long long gid = (long long)blockIdx.x * blockDim.x + threadIdx.x;
    long long total = (long long)E << lgChunks;
    if (gid >= total) return;
    int e  = (int)(gid >> lgChunks);
    int c4 = ((int)gid & ((1 << lgChunks) - 1)) << 2;
    int s = src[e];
    int d = dst[e];
    float coef = dinv[s] * ew[e] * dinv[d];
    const float4 v = *reinterpret_cast<const float4*>(h + (size_t)s * C + c4);
    float* a = acc + (size_t)d * C + c4;
    atomicAdd(a + 0, v.x * coef);
    atomicAdd(a + 1, v.y * coef);
    atomicAdd(a + 2, v.z * coef);
    atomicAdd(a + 3, v.w * coef);
}

// out[i] = acc[i] + bias[i & (C-1)]  (optional ReLU)
__global__ void finalize(const float* __restrict__ acc, const float* __restrict__ bias,
                         float* __restrict__ out, int total, int mask, int relu) {
    int i = blockIdx.x * blockDim.x + threadIdx.x;
    if (i < total) {
        float v = acc[i] + bias[i & mask];
        out[i] = (relu && v < 0.0f) ? 0.0f : v;
    }
}

// ---------------------------------------------------------------------------
// Launch
// ---------------------------------------------------------------------------
static inline int cdiv(long long a, long long b) { return (int)((a + b - 1) / b); }

extern "C" void kernel_launch(void* const* d_in, const int* in_sizes, int n_in,
                              void* d_out, int out_size, void* d_ws, size_t ws_size,
                              hipStream_t stream) {
    const float* user = (const float*)d_in[0];
    const float* item = (const float*)d_in[1];
    const int*   ei   = (const int*)d_in[2];     // [2, E]
    const float* ew   = (const float*)d_in[3];
    const float* Wu   = (const float*)d_in[4];
    const float* bu   = (const float*)d_in[5];
    const float* Wi   = (const float*)d_in[6];
    const float* bi   = (const float*)d_in[7];
    const float* W1   = (const float*)d_in[8];
    const float* b1   = (const float*)d_in[9];
    const float* W2   = (const float*)d_in[10];
    const float* b2   = (const float*)d_in[11];
    const int* srcIdx = ei;
    const int* dstIdx = ei + N_EDGES;
    float* out = (float*)d_out;

    // Workspace carve-up (256B aligned slices)
    char* base = (char*)d_ws;
    size_t off = 0;
    auto carve = [&](size_t bytes) -> char* {
        char* p = base + off;
        off += (bytes + 255) & ~(size_t)255;
        return p;
    };
    float*    x0   = (float*)carve((size_t)N_NODES * COMMON * sizeof(float));
    float*    h    = (float*)carve((size_t)N_NODES * COMMON * sizeof(float));
    float*    acc  = (float*)carve((size_t)N_NODES * COMMON * sizeof(float));
    float*    deg  = (float*)carve((size_t)N_NODES * sizeof(float));
    _Float16* WuP  = (_Float16*)carve((size_t)USER_DIM * COMMON * sizeof(_Float16));
    _Float16* WiP  = (_Float16*)carve((size_t)ITEM_DIM * COMMON * sizeof(_Float16));
    _Float16* W1P  = (_Float16*)carve((size_t)COMMON * HIDDEN * sizeof(_Float16));
    _Float16* W2P  = (_Float16*)carve((size_t)HIDDEN * OUT_DIM * sizeof(_Float16));
    (void)ws_size; (void)in_sizes; (void)n_in; (void)out_size;

    // 1) Pack weights into WMMA B-fragment layout (tiny kernels)
    pack_wmma_b<<<(COMMON/16)*(USER_DIM/32), 32, 0, stream>>>(Wu, WuP, COMMON, USER_DIM/32);
    pack_wmma_b<<<(COMMON/16)*(ITEM_DIM/32), 32, 0, stream>>>(Wi, WiP, COMMON, ITEM_DIM/32);
    pack_wmma_b<<<(HIDDEN/16)*(COMMON/32),   32, 0, stream>>>(W1, W1P, HIDDEN, COMMON/32);
    pack_wmma_b<<<(OUT_DIM/16)*(HIDDEN/32),  32, 0, stream>>>(W2, W2P, OUT_DIM, HIDDEN/32);

    // 2) Projections -> x0 = concat(user@Wu+bu, item@Wi+bi)  [N_NODES, 64]
    wmma_gemm_f16<4><<<cdiv(N_USERS / 16, 8), 256, 0, stream>>>(
        user, WuP, bu, x0, N_USERS, USER_DIM);
    wmma_gemm_f16<4><<<cdiv(N_ITEMS / 16, 8), 256, 0, stream>>>(
        item, WiP, bi, x0 + (size_t)N_USERS * COMMON, N_ITEMS, ITEM_DIM);

    // 3) Degree with self-loops -> dinv (in place)
    fill_f32<<<cdiv(N_NODES, 256), 256, 0, stream>>>(deg, 1.0f, N_NODES);
    deg_scatter<<<cdiv(N_EDGES, 256), 256, 0, stream>>>(dstIdx, ew, deg, N_EDGES);
    rsqrt_inplace<<<cdiv(N_NODES, 256), 256, 0, stream>>>(deg, N_NODES);
    float* dinv = deg;

    // 4) Layer 1: h = x0 @ W1 ; acc = self + edge messages ; x1 = relu(acc+b1)
    {
        wmma_gemm_f16<4><<<cdiv(N_NODES / 16, 8), 256, 0, stream>>>(
            x0, W1P, nullptr, h, N_NODES, COMMON);
        int total = N_NODES * HIDDEN;
        self_msg<<<cdiv(total, 256), 256, 0, stream>>>(h, dinv, acc, total, 6);
        long long work = (long long)N_EDGES * (HIDDEN / 4);
        edge_scatter<<<cdiv(work, 256), 256, 0, stream>>>(
            h, srcIdx, dstIdx, ew, dinv, acc, N_EDGES, HIDDEN, 4);
        finalize<<<cdiv(total, 256), 256, 0, stream>>>(
            acc, b1, x0, total, HIDDEN - 1, 1);   // x1 overwrites x0
    }

    // 5) Layer 2: h2 = x1 @ W2 ; acc2 = self + edge ; out = acc2 + b2
    {
        wmma_gemm_f16<2><<<cdiv(N_NODES / 16, 8), 256, 0, stream>>>(
            x0, W2P, nullptr, h, N_NODES, HIDDEN);
        int total = N_NODES * OUT_DIM;
        self_msg<<<cdiv(total, 256), 256, 0, stream>>>(h, dinv, acc, total, 5);
        long long work = (long long)N_EDGES * (OUT_DIM / 4);
        edge_scatter<<<cdiv(work, 256), 256, 0, stream>>>(
            h, srcIdx, dstIdx, ew, dinv, acc, N_EDGES, OUT_DIM, 3);
        finalize<<<cdiv(total, 256), 256, 0, stream>>>(
            acc, b2, out, total, OUT_DIM - 1, 0);
    }
}
